// MultiHeadAttention_60936995996017
// MI455X (gfx1250) — matrix operand order repaired and checked by reading, not compiled, not run on hardware
//
#include <hip/hip_runtime.h>
#include <hip/hip_bf16.h>

#define S_LEN 2048
#define D_MODEL 1024
#define NHEAD 16
#define DHEAD 64
#define BATCH 2

typedef __attribute__((ext_vector_type(16))) _Float16 v16h;
typedef __attribute__((ext_vector_type(8)))  _Float16 v8h;
typedef __attribute__((ext_vector_type(8)))  float    v8f;

union F16x16 { v16h v; v8h h8[2]; };

__device__ __forceinline__ v8f wmma_f16(v16h a, v16h b, v8f c) {
  // D = A(16x32 f16) * B(32x16 f16) + C(16x16 f32)
  return __builtin_amdgcn_wmma_f32_16x16x32_f16(false, a, false, b, (short)0, c, false, false);
}

// A-matrix 16x32 f16 fragment: lane&15 = row M; lanes<16 hold
// K=0..15; lanes>=16 hold K=16..31 (contiguous, matching B).
// => per lane: one contiguous 16-half (32B) run at k0+(lane>>4)*16.
__device__ __forceinline__ v16h load_a_frag(const _Float16* base, int rowBase, int ld, int k0, int lane) {
  const _Float16* p = base + (size_t)(rowBase + (lane & 15)) * ld + k0 + ((lane >> 4) << 4);
  F16x16 u;
  u.h8[0] = *(const v8h*)(p);
  u.h8[1] = *(const v8h*)(p + 8);
  return u.v;
}

// B-matrix 32x16 f16 fragment: lane&15 = column N; lanes<16 hold K=0..15,
// lanes>=16 hold K=16..31 (contiguous). Memory holds B^T row-major
// (row = column of B, contiguous over K) => one 32B contiguous run per lane.
__device__ __forceinline__ v16h load_b_frag(const _Float16* baseT, int colBase, int ld, int k0, int lane) {
  const _Float16* p = baseT + (size_t)(colBase + (lane & 15)) * ld + k0 + ((lane >> 4) << 4);
  F16x16 u;
  u.h8[0] = *(const v8h*)(p);
  u.h8[1] = *(const v8h*)(p + 8);
  return u.v;
}

// ---------------- elementwise f32 -> f16 ----------------
__global__ __launch_bounds__(256) void cvt_f32_f16(const float* __restrict__ in,
                                                   _Float16* __restrict__ out, int n) {
  int i = blockIdx.x * 256 + threadIdx.x;
  if (i < n) out[i] = (_Float16)in[i];
}

// ------------- W[D,D] f32 -> Wt[D,D] f16 (Wt[n][k] = W[k][n]) -------------
__global__ __launch_bounds__(256) void transpose_w(const float* __restrict__ W,
                                                   _Float16* __restrict__ Wt) {
  __shared__ float tile[32][33];
  int bx = blockIdx.x * 32;  // k
  int by = blockIdx.y * 32;  // n
  int tx = threadIdx.x;      // 0..31
  int ty = threadIdx.y;      // 0..7
#pragma unroll
  for (int i = 0; i < 32; i += 8)
    tile[ty + i][tx] = W[(size_t)(bx + ty + i) * D_MODEL + (by + tx)];
  __syncthreads();
#pragma unroll
  for (int i = 0; i < 32; i += 8)
    Wt[(size_t)(by + ty + i) * D_MODEL + (bx + tx)] = (_Float16)tile[tx][ty + i];
}

// ---------------- GEMM: C[4096,1024] = A(f16) @ Bt^T(f16) + bias ----------------
// mode 0: store f16 head-layout [B,H,S,DH]     (Q, K projections)
// mode 1: store f16 transposed  [B,H,DH,S]     (V projection)
// mode 2: store f32 row-major   [B*S, D]       (output projection -> d_out)
__device__ __forceinline__ void store_tile(v8f c, int mT, int nT, const float* bias,
                                           void* outp, int mode, int lane) {
#pragma unroll
  for (int r = 0; r < 8; ++r) {
    int m = mT + ((lane >> 4) << 3) + r;   // C layout: vgpr r -> row r / 8+r
    int n = nT + (lane & 15);              //           lane&15 -> column
    float val = c[r] + bias[n];
    if (mode == 2) {
      ((float*)outp)[(size_t)m * D_MODEL + n] = val;
    } else {
      int b = m >> 11, s = m & (S_LEN - 1);
      int h = n >> 6,  d = n & (DHEAD - 1);
      _Float16* o16 = (_Float16*)outp;
      if (mode == 0)
        o16[(((size_t)b * NHEAD + h) * S_LEN + s) * DHEAD + d] = (_Float16)val;
      else
        o16[(((size_t)b * NHEAD + h) * DHEAD + d) * S_LEN + s] = (_Float16)val;
    }
  }
}

// 4 waves (2x2), each wave computes a 64x64 tile: 16 WMMA accumulators,
// 8 fragment loads per K-step -> 1 b128 load per WMMA.
__global__ __launch_bounds__(128) void gemm_kernel(const _Float16* __restrict__ A,
                                                   const _Float16* __restrict__ Bt,
                                                   const float* __restrict__ bias,
                                                   void* __restrict__ outp, int mode) {
  const int lane = threadIdx.x & 31;
  const int wave = threadIdx.x >> 5;                      // 4 waves, 2x2
  const int mBase = blockIdx.x * 128 + (wave & 1) * 64;   // 128-row block tile
  const int nBase = blockIdx.y * 128 + (wave >> 1) * 64;  // 128-col block tile
  v8f c[4][4] = {};
  for (int k0 = 0; k0 < D_MODEL; k0 += 32) {
    v16h a0 = load_a_frag(A, mBase,      D_MODEL, k0, lane);
    v16h a1 = load_a_frag(A, mBase + 16, D_MODEL, k0, lane);
    v16h a2 = load_a_frag(A, mBase + 32, D_MODEL, k0, lane);
    v16h a3 = load_a_frag(A, mBase + 48, D_MODEL, k0, lane);
#pragma unroll
    for (int j = 0; j < 4; ++j) {
      v16h b = load_b_frag(Bt, nBase + 16 * j, D_MODEL, k0, lane);
      c[0][j] = wmma_f16(a0, b, c[0][j]);
      c[1][j] = wmma_f16(a1, b, c[1][j]);
      c[2][j] = wmma_f16(a2, b, c[2][j]);
      c[3][j] = wmma_f16(a3, b, c[3][j]);
    }
  }
#pragma unroll
  for (int i = 0; i < 4; ++i)
#pragma unroll
    for (int j = 0; j < 4; ++j)
      store_tile(c[i][j], mBase + 16 * i, nBase + 16 * j, bias, outp, mode, lane);
}

// ---------------- flash attention: one wave per 16-query tile per (b,h) ----------------
__global__ __launch_bounds__(32) void attn_kernel(const _Float16* __restrict__ Qh,
                                                  const _Float16* __restrict__ Kh,
                                                  const _Float16* __restrict__ Vt,
                                                  const float* __restrict__ rel_bias,
                                                  _Float16* __restrict__ Aout) {
  __shared__ __align__(16) _Float16 pbuf[16 * 32];
  const int lane = threadIdx.x & 31;
  const int q0 = blockIdx.x * 16;
  const int h  = blockIdx.y;
  const int b  = blockIdx.z;

  const _Float16* Qbh = Qh + ((size_t)b * NHEAD + h) * S_LEN * DHEAD;
  const _Float16* Kbh = Kh + ((size_t)b * NHEAD + h) * S_LEN * DHEAD;
  const _Float16* Vbh = Vt + ((size_t)b * NHEAD + h) * DHEAD * S_LEN;

  v16h qf0 = load_a_frag(Qbh, q0, DHEAD, 0, lane);
  v16h qf1 = load_a_frag(Qbh, q0, DHEAD, 32, lane);

  v8f o0 = {}, o1 = {}, o2 = {}, o3 = {};
  float rowMax[8], rowSum[8];
#pragma unroll
  for (int r = 0; r < 8; ++r) { rowMax[r] = -__builtin_inff(); rowSum[r] = 0.f; }
  const float scale = 0.125f;  // 1/sqrt(64)

  for (int kc = 0; kc < q0 + 16; kc += 32) {  // causal: keys 0..q0+15
    v16h kf00 = load_b_frag(Kbh, kc,      DHEAD, 0,  lane);
    v16h kf01 = load_b_frag(Kbh, kc,      DHEAD, 32, lane);
    v16h kf10 = load_b_frag(Kbh, kc + 16, DHEAD, 0,  lane);
    v16h kf11 = load_b_frag(Kbh, kc + 16, DHEAD, 32, lane);
    v8f s0 = {}, s1 = {};
    s0 = wmma_f16(qf0, kf00, s0);
    s0 = wmma_f16(qf1, kf01, s0);
    s1 = wmma_f16(qf0, kf10, s1);
    s1 = wmma_f16(qf1, kf11, s1);

    const bool moreChunks = (kc + 32) < (q0 + 16);
    float alphaArr[8];
#pragma unroll
    for (int r = 0; r < 8; ++r) {
      int lrow = ((lane >> 4) << 3) + r;      // row within tile
      int qIdx = q0 + lrow;
      int key0 = kc + (lane & 15);
      int key1 = key0 + 16;
      const float* rb = rel_bias + ((size_t)h * S_LEN + qIdx) * S_LEN;
      // hide HBM latency of the dominant rel_bias stream behind this chunk's WMMAs
      if (moreChunks) {
        __builtin_prefetch(rb + key0 + 32, 0, 1);
        __builtin_prefetch(rb + key1 + 32, 0, 1);
      }
      float x0 = s0[r] * scale + rb[key0];
      float x1 = s1[r] * scale + rb[key1];
      x0 = (key0 <= qIdx) ? x0 : -__builtin_inff();
      x1 = (key1 <= qIdx) ? x1 : -__builtin_inff();
      // row max across the 16 lanes of this half-wave
      float m = fmaxf(x0, x1);
      m = fmaxf(m, __shfl_xor(m, 1, 32));
      m = fmaxf(m, __shfl_xor(m, 2, 32));
      m = fmaxf(m, __shfl_xor(m, 4, 32));
      m = fmaxf(m, __shfl_xor(m, 8, 32));
      float nm = fmaxf(rowMax[r], m);
      float alpha = __expf(rowMax[r] - nm);
      float p0 = __expf(x0 - nm);
      float p1 = __expf(x1 - nm);
      float ps = p0 + p1;
      ps += __shfl_xor(ps, 1, 32);
      ps += __shfl_xor(ps, 2, 32);
      ps += __shfl_xor(ps, 4, 32);
      ps += __shfl_xor(ps, 8, 32);
      rowSum[r] = rowSum[r] * alpha + ps;
      rowMax[r] = nm;
      alphaArr[r] = alpha;
      // stage P to LDS row-major [16 x 32] for A-fragment reload
      pbuf[lrow * 32 + (lane & 15)]      = (_Float16)p0;
      pbuf[lrow * 32 + 16 + (lane & 15)] = (_Float16)p1;
    }
#pragma unroll
    for (int r = 0; r < 8; ++r) {
      o0[r] *= alphaArr[r]; o1[r] *= alphaArr[r];
      o2[r] *= alphaArr[r]; o3[r] *= alphaArr[r];
    }
    __syncthreads();  // LDS stores -> loads fence (single wave, DS in order)
    v16h pf  = load_a_frag(pbuf, 0, 32, 0, lane);
    v16h vf0 = load_b_frag(Vbh, 0,  S_LEN, kc, lane);
    v16h vf1 = load_b_frag(Vbh, 16, S_LEN, kc, lane);
    v16h vf2 = load_b_frag(Vbh, 32, S_LEN, kc, lane);
    v16h vf3 = load_b_frag(Vbh, 48, S_LEN, kc, lane);
    o0 = wmma_f16(pf, vf0, o0);
    o1 = wmma_f16(pf, vf1, o1);
    o2 = wmma_f16(pf, vf2, o2);
    o3 = wmma_f16(pf, vf3, o3);
    __syncthreads();
  }

#pragma unroll
  for (int r = 0; r < 8; ++r) {
    float inv = 1.0f / rowSum[r];
    int s = q0 + ((lane >> 4) << 3) + r;
    _Float16* row = Aout + ((size_t)b * S_LEN + s) * D_MODEL + h * DHEAD;
    row[ 0 + (lane & 15)] = (_Float16)(o0[r] * inv);
    row[16 + (lane & 15)] = (_Float16)(o1[r] * inv);
    row[32 + (lane & 15)] = (_Float16)(o2[r] * inv);
    row[48 + (lane & 15)] = (_Float16)(o3[r] * inv);
  }
}

extern "C" void kernel_launch(void* const* d_in, const int* in_sizes, int n_in,
                              void* d_out, int out_size, void* d_ws, size_t ws_size,
                              hipStream_t stream) {
  (void)in_sizes; (void)n_in; (void)out_size; (void)ws_size;
  const float* q  = (const float*)d_in[0];
  const float* k  = (const float*)d_in[1];
  const float* v  = (const float*)d_in[2];
  /* d_in[3] = attn_mask: causal tril, reproduced by index compare */
  const float* rel_bias = (const float*)d_in[4];
  const float* Wq = (const float*)d_in[5];  const float* bq = (const float*)d_in[6];
  const float* Wk = (const float*)d_in[7];  const float* bk = (const float*)d_in[8];
  const float* Wv = (const float*)d_in[9];  const float* bv = (const float*)d_in[10];
  const float* Wo = (const float*)d_in[11]; const float* bo = (const float*)d_in[12];
  float* out = (float*)d_out;

  const int    NACT   = BATCH * S_LEN * D_MODEL;          // 4,194,304 elems
  const size_t SZ_ACT = (size_t)NACT * sizeof(_Float16);  // 8 MB
  const size_t SZ_W   = (size_t)D_MODEL * D_MODEL * sizeof(_Float16);  // 2 MB

  char* w = (char*)d_ws;                    // 64 MB total
  _Float16* qf   = (_Float16*)w; w += SZ_ACT;
  _Float16* kf   = (_Float16*)w; w += SZ_ACT;
  _Float16* vf   = (_Float16*)w; w += SZ_ACT;
  _Float16* Wqt  = (_Float16*)w; w += SZ_W;
  _Float16* Wkt  = (_Float16*)w; w += SZ_W;
  _Float16* Wvt  = (_Float16*)w; w += SZ_W;
  _Float16* Wot  = (_Float16*)w; w += SZ_W;
  _Float16* Qh   = (_Float16*)w; w += SZ_ACT;
  _Float16* Kh   = (_Float16*)w; w += SZ_ACT;
  _Float16* Vtw  = (_Float16*)w; w += SZ_ACT;
  _Float16* Aout = (_Float16*)w; w += SZ_ACT;

  dim3 cvtGrid((NACT + 255) / 256);
  cvt_f32_f16<<<cvtGrid, 256, 0, stream>>>(q, qf, NACT);
  cvt_f32_f16<<<cvtGrid, 256, 0, stream>>>(k, kf, NACT);
  cvt_f32_f16<<<cvtGrid, 256, 0, stream>>>(v, vf, NACT);

  dim3 tGrid(D_MODEL / 32, D_MODEL / 32), tBlk(32, 8);
  transpose_w<<<tGrid, tBlk, 0, stream>>>(Wq, Wqt);
  transpose_w<<<tGrid, tBlk, 0, stream>>>(Wk, Wkt);
  transpose_w<<<tGrid, tBlk, 0, stream>>>(Wv, Wvt);
  transpose_w<<<tGrid, tBlk, 0, stream>>>(Wo, Wot);

  dim3 gGrid(BATCH * S_LEN / 128, D_MODEL / 128);  // (32, 8)
  gemm_kernel<<<gGrid, 128, 0, stream>>>(qf, Wqt, bq, Qh, 0);
  gemm_kernel<<<gGrid, 128, 0, stream>>>(kf, Wkt, bk, Kh, 0);
  gemm_kernel<<<gGrid, 128, 0, stream>>>(vf, Wvt, bv, Vtw, 1);

  dim3 aGrid(S_LEN / 16, NHEAD, BATCH);  // (128, 16, 2)
  attn_kernel<<<aGrid, 32, 0, stream>>>(Qh, Kh, Vtw, rel_bias, Aout);

  gemm_kernel<<<gGrid, 128, 0, stream>>>(Aout, Wot, bo, out, 2);
}
